// MultiHeadAttention_24240795419224
// MI455X (gfx1250) — matrix operand order repaired and checked by reading, not compiled, not run on hardware
//
#include <hip/hip_runtime.h>

#define Bc 2
#define Sc 2048
#define Dc 1024
#define Hc 16
#define HDc 64
#define Mc (Bc*Sc)   // 4096

typedef __attribute__((ext_vector_type(16))) __bf16 v16bf;
typedef __attribute__((ext_vector_type(8)))  float  v8f;

union FragB {
  v16bf          v;
  unsigned short u[16];
  unsigned int   w[8];
  uint4          q[2];
};
union AccF {
  v8f   v;
  float f[8];
};

__device__ __forceinline__ unsigned short f2bf(float x) {
  unsigned int uu = __float_as_uint(x);
  uu = (uu + 0x7FFFu + ((uu >> 16) & 1u)) >> 16;
  return (unsigned short)uu;
}

// ---- CDNA5 async global->LDS copy (ASYNCcnt-tracked), per ISA 10.7/15.18 ----
__device__ __forceinline__ void async_ld_b128(unsigned lds_off, const void* g) {
  asm volatile("global_load_async_to_lds_b128 %0, %1, off"
               :: "v"(lds_off), "v"((unsigned long long)(uintptr_t)g)
               : "memory");
}
__device__ __forceinline__ void wait_async0() {
  asm volatile("s_wait_asynccnt 0x0" ::: "memory");
}
__device__ __forceinline__ void wait_async1() {
  asm volatile("s_wait_asynccnt 0x1" ::: "memory");
}

// ---- 16-lane xor-butterfly max (VALU permlane if available, else shuffles) ----
__device__ __forceinline__ float rowmax16(float x) {
#if __has_builtin(__builtin_amdgcn_permlane16)
#define PL16(v, lo, hi) __uint_as_float(__builtin_amdgcn_permlane16( \
    __float_as_uint(v), __float_as_uint(v), lo, hi, false, false))
  x = fmaxf(x, PL16(x, 0x67452301u, 0xEFCDAB89u));  // xor 1
  x = fmaxf(x, PL16(x, 0x54761032u, 0xDCFE98BAu));  // xor 2
  x = fmaxf(x, PL16(x, 0x32107654u, 0xBA98FEDCu));  // xor 4
  x = fmaxf(x, PL16(x, 0xFEDCBA98u, 0x76543210u));  // xor 8
#undef PL16
#else
  #pragma unroll
  for (int off = 1; off < 16; off <<= 1)
    x = fmaxf(x, __shfl_xor(x, off, 16));
#endif
  return x;
}

__global__ void cvt_f32_bf16(const float* __restrict__ in,
                             unsigned short* __restrict__ out, int n) {
  int i = blockIdx.x * blockDim.x + threadIdx.x;
  if (i < n) out[i] = f2bf(in[i]);
}

// C[M,N] = A[M,K] * B[K,N], K=N=1024, M=4096, bf16 in, f32 acc.
// Double-buffered LDS; A tile staged with async-to-LDS, B transposed via regs.
// MODE 0: bf16 out [B,H,S,HD] (Q,K)  MODE 1: bf16 out [B,H,HD,S] (V^T)
// MODE 2: f32 out row-major + bias (final projection)
template<int MODE>
__global__ __launch_bounds__(256)
void gemm_wmma(const unsigned short* __restrict__ A,
               const unsigned short* __restrict__ Bm,
               void* __restrict__ outp,
               const float* __restrict__ bias) {
  __shared__ unsigned short Alds[2][64 * 32];   // [m][k] row-major
  __shared__ unsigned short Blds[2][128 * 32];  // [n][k] (transposed at store)

  const int tid  = threadIdx.x;
  const int lane = tid & 31;
  const int wave = tid >> 5;
  const int wy   = wave & 1;    // M: 2 waves * 32
  const int wx   = wave >> 1;   // N: 4 waves * 32
  const int col  = lane & 15;
  const int hi   = lane >> 4;
  const int m0   = blockIdx.y * 64;
  const int n0   = blockIdx.x * 128;

  AccF c[2][2];
  #pragma unroll
  for (int i = 0; i < 2; i++)
    #pragma unroll
    for (int j = 0; j < 2; j++)
      #pragma unroll
      for (int r = 0; r < 8; r++) c[i][j].f[r] = 0.0f;

  const int arow = tid >> 2;          // 0..63
  const int acol = (tid & 3) * 8;     // 0,8,16,24
  const int bk   = tid >> 3;          // 0..31
  const int bn   = (tid & 7) * 16;    // 0..112

  const unsigned short* Abase = A + (size_t)(m0 + arow) * Dc + acol;
  const unsigned short* Bbase = Bm + (size_t)bk * Dc + n0 + bn;
  const unsigned aoff0 = (unsigned)(uintptr_t)&Alds[0][arow * 32 + acol];
  const unsigned aoff1 = (unsigned)(uintptr_t)&Alds[1][arow * 32 + acol];

  // prologue: chunk 0 in flight
  uint4 bv0 = *(const uint4*)(Bbase);
  uint4 bv1 = *(const uint4*)(Bbase + 8);
  async_ld_b128(aoff0, Abase);

  for (int kc = 0; kc < Dc; kc += 32) {
    const int p = (kc >> 5) & 1;
    // 1) publish B(kc) into Blds[p]
    {
      const unsigned short* pb = (const unsigned short*)&bv0;
      #pragma unroll
      for (int i = 0; i < 8; i++) Blds[p][(bn + i) * 32 + bk] = pb[i];
      pb = (const unsigned short*)&bv1;
      #pragma unroll
      for (int i = 0; i < 8; i++) Blds[p][(bn + 8 + i) * 32 + bk] = pb[i];
    }
    // 2) issue next chunk; 3) wait for this chunk's async A
    if (kc + 32 < Dc) {
      async_ld_b128(p ? aoff0 : aoff1, Abase + kc + 32);
      bv0 = *(const uint4*)(Bbase + (size_t)(kc + 32) * Dc);
      bv1 = *(const uint4*)(Bbase + (size_t)(kc + 32) * Dc + 8);
      wait_async1();
    } else {
      wait_async0();
    }
    __syncthreads();   // buffer p fully valid for every wave

    FragB a[2], b[2];
    #pragma unroll
    for (int i = 0; i < 2; i++) {
      int m = wy * 32 + i * 16 + col;
      #pragma unroll
      for (int v = 0; v < 8; v++) {
        int kk = hi * 16 + v * 2;
        a[i].w[v] = *(const unsigned int*)(&Alds[p][m * 32 + kk]);
      }
    }
    #pragma unroll
    for (int j = 0; j < 2; j++) {
      int n = wx * 32 + j * 16 + col;
      b[j].q[0] = *(const uint4*)(&Blds[p][n * 32 + hi * 16]);
      b[j].q[1] = *(const uint4*)(&Blds[p][n * 32 + hi * 16 + 8]);
    }
    #pragma unroll
    for (int i = 0; i < 2; i++)
      #pragma unroll
      for (int j = 0; j < 2; j++)
        c[i][j].v = __builtin_amdgcn_wmma_f32_16x16x32_bf16(
            false, a[i].v, false, b[j].v, (short)0, c[i][j].v, false, false);

    __syncthreads();   // reads of buffer p done before it is refilled
  }

  #pragma unroll
  for (int i = 0; i < 2; i++) {
    #pragma unroll
    for (int j = 0; j < 2; j++) {
      #pragma unroll
      for (int r = 0; r < 8; r++) {
        int m = m0 + wy * 32 + i * 16 + hi * 8 + r;
        int n = n0 + wx * 32 + j * 16 + col;
        float val = c[i][j].f[r];
        if (MODE == 2) {
          ((float*)outp)[(size_t)m * Dc + n] = val + bias[n];
        } else {
          int bb = m / Sc, s = m % Sc;
          int h  = n / HDc, hd = n % HDc;
          size_t addr;
          if (MODE == 0) addr = (((size_t)(bb * Hc + h) * Sc) + s) * HDc + hd;
          else           addr = (((size_t)(bb * Hc + h) * HDc) + hd) * Sc + s;
          ((unsigned short*)outp)[addr] = f2bf(val);
        }
      }
    }
  }
}

// One wave per 16-query tile. Q,K: [B,H,S,HD] bf16; VT: [B,H,HD,S] bf16.
// Flash attention, causal, unscaled. Row-sum via P@ones WMMA; row-max via permlane.
__global__ __launch_bounds__(32)
void attn_kernel(const unsigned short* __restrict__ Q,
                 const unsigned short* __restrict__ K,
                 const unsigned short* __restrict__ VT,
                 unsigned short* __restrict__ CTX) {
  __shared__ unsigned short Plds[16 * 32];

  const int lane = threadIdx.x & 31;
  const int col  = lane & 15;
  const int hi   = lane >> 4;
  const int blk  = blockIdx.x;
  const int qt   = blk & 127;        // S/16 tiles
  const int h    = (blk >> 7) & 15;
  const int bb   = blk >> 11;
  const int q0   = qt * 16;
  const size_t bhS = (size_t)(bb * Hc + h) * Sc;

  // Q A-fragments for the two K-dim chunks (HD = 64 = 2*32)
  FragB aQ[2];
  {
    const unsigned short* qrow = Q + (bhS + q0 + col) * HDc;
    #pragma unroll
    for (int cc = 0; cc < 2; cc++)
      #pragma unroll
      for (int v = 0; v < 8; v++) {
        int kk = cc * 32 + hi * 16 + v * 2;
        aQ[cc].w[v] = *(const unsigned int*)(qrow + kk);
      }
  }
  FragB ones;  // B-fragment of all 1.0bf16 -> rowsum WMMA
  #pragma unroll
  for (int i = 0; i < 8; i++) ones.w[i] = 0x3F803F80u;

  AccF acc[4], lacc;
  #pragma unroll
  for (int nt = 0; nt < 4; nt++)
    #pragma unroll
    for (int r = 0; r < 8; r++) acc[nt].f[r] = 0.0f;
  #pragma unroll
  for (int r = 0; r < 8; r++) lacc.f[r] = 0.0f;
  float mrow[8];
  #pragma unroll
  for (int r = 0; r < 8; r++) mrow[r] = -1e30f;

  const int jEnd = q0 + 16;
  for (int j = 0; j < jEnd; j += 32) {
    // ---- scores: two 16x16 tiles over 32 keys, K-dim = HD via 2 chained WMMAs
    AccF s[2];
    #pragma unroll
    for (int t = 0; t < 2; t++) {
      int key = j + t * 16 + col;
      if (key > Sc - 1) key = Sc - 1;            // clamp; masked below anyway
      const unsigned short* krow = K + (bhS + key) * HDc;
      FragB bk0, bk1;
      bk0.q[0] = *(const uint4*)(krow + hi * 16);
      bk0.q[1] = *(const uint4*)(krow + hi * 16 + 8);
      bk1.q[0] = *(const uint4*)(krow + 32 + hi * 16);
      bk1.q[1] = *(const uint4*)(krow + 32 + hi * 16 + 8);
      AccF z;
      #pragma unroll
      for (int r = 0; r < 8; r++) z.f[r] = 0.0f;
      s[t].v = __builtin_amdgcn_wmma_f32_16x16x32_bf16(
          false, aQ[0].v, false, bk0.v, (short)0, z.v, false, false);
      s[t].v = __builtin_amdgcn_wmma_f32_16x16x32_bf16(
          false, aQ[1].v, false, bk1.v, (short)0, s[t].v, false, false);
    }
    // ---- causal mask + online softmax (rows r+8*hi; cols = lane%16)
    #pragma unroll
    for (int r = 0; r < 8; r++) {
      int qrow_i = q0 + hi * 8 + r;
      float v0 = ((j + col)      <= qrow_i) ? s[0].f[r] : -1e30f;
      float v1 = ((j + 16 + col) <= qrow_i) ? s[1].f[r] : -1e30f;
      float mx = rowmax16(fmaxf(v0, v1));
      float mnew = fmaxf(mrow[r], mx);
      float corr = __expf(mrow[r] - mnew);
      mrow[r] = mnew;
      s[0].f[r] = __expf(v0 - mnew);
      s[1].f[r] = __expf(v1 - mnew);
      lacc.f[r] *= corr;
      #pragma unroll
      for (int nt = 0; nt < 4; nt++) acc[nt].f[r] *= corr;
    }
    // ---- P: C-layout -> A-layout via LDS (bf16)
    __syncthreads();
    #pragma unroll
    for (int r = 0; r < 8; r++) {
      Plds[(hi * 8 + r) * 32 + col]      = f2bf(s[0].f[r]);
      Plds[(hi * 8 + r) * 32 + 16 + col] = f2bf(s[1].f[r]);
    }
    __syncthreads();
    FragB aP;
    #pragma unroll
    for (int v = 0; v < 8; v++) {
      int kk = hi * 16 + v * 2;
      aP.w[v] = *(const unsigned int*)(Plds + col * 32 + kk);
    }
    __syncthreads();
    // ---- l += rowsum(P) via P @ ones
    lacc.v = __builtin_amdgcn_wmma_f32_16x16x32_bf16(
        false, aP.v, false, ones.v, (short)0, lacc.v, false, false);
    // ---- acc += P @ V  (B-frags contiguous thanks to VT layout)
    int kstart = j + hi * 16;
    if (kstart > Sc - 16) kstart = Sc - 16;      // only hit when fully masked
    #pragma unroll
    for (int nt = 0; nt < 4; nt++) {
      const unsigned short* vrow =
          VT + ((size_t)(bb * Hc + h) * HDc + nt * 16 + col) * Sc + kstart;
      FragB bv;
      bv.q[0] = *(const uint4*)(vrow);
      bv.q[1] = *(const uint4*)(vrow + 8);
      acc[nt].v = __builtin_amdgcn_wmma_f32_16x16x32_bf16(
          false, aP.v, false, bv.v, (short)0, acc[nt].v, false, false);
    }
  }

  // ---- normalize + store ctx as bf16 row-major [B*S, D] (n = h*64+hd)
  #pragma unroll
  for (int r = 0; r < 8; r++) {
    float inv = 1.0f / lacc.f[r];
    size_t m = (size_t)bb * Sc + q0 + hi * 8 + r;
    #pragma unroll
    for (int nt = 0; nt < 4; nt++)
      CTX[m * Dc + h * HDc + nt * 16 + col] = f2bf(acc[nt].f[r] * inv);
  }
}

extern "C" void kernel_launch(void* const* d_in, const int* in_sizes, int n_in,
                              void* d_out, int out_size, void* d_ws, size_t ws_size,
                              hipStream_t stream) {
  const float* X  = (const float*)d_in[0];
  const float* Wq = (const float*)d_in[1];
  const float* Wk = (const float*)d_in[2];
  const float* Wv = (const float*)d_in[3];
  const float* Wo = (const float*)d_in[4];
  const float* bo = (const float*)d_in[5];

  unsigned short* ws  = (unsigned short*)d_ws;
  unsigned short* Xb  = ws;                          // [M,K]  bf16
  unsigned short* Wqb = Xb  + (size_t)Mc * Dc;
  unsigned short* Wkb = Wqb + (size_t)Dc * Dc;
  unsigned short* Wvb = Wkb + (size_t)Dc * Dc;
  unsigned short* Wob = Wvb + (size_t)Dc * Dc;
  unsigned short* Qb  = Wob + (size_t)Dc * Dc;       // [B,H,S,HD]
  unsigned short* Kb  = Qb  + (size_t)Mc * Dc;       // [B,H,S,HD]
  unsigned short* VTb = Kb  + (size_t)Mc * Dc;       // [B,H,HD,S]
  unsigned short* CTXb= VTb + (size_t)Mc * Dc;       // [M,D]
  (void)ws_size; (void)in_sizes; (void)n_in; (void)out_size;

  const int nX = Mc * Dc;
  const int nW = Dc * Dc;
  cvt_f32_bf16<<<(nX + 255) / 256, 256, 0, stream>>>(X,  Xb,  nX);
  cvt_f32_bf16<<<(nW + 255) / 256, 256, 0, stream>>>(Wq, Wqb, nW);
  cvt_f32_bf16<<<(nW + 255) / 256, 256, 0, stream>>>(Wk, Wkb, nW);
  cvt_f32_bf16<<<(nW + 255) / 256, 256, 0, stream>>>(Wv, Wvb, nW);
  cvt_f32_bf16<<<(nW + 255) / 256, 256, 0, stream>>>(Wo, Wob, nW);

  dim3 gg(Dc / 128, Mc / 64);   // (8, 64)
  gemm_wmma<0><<<gg, 256, 0, stream>>>(Xb, Wqb, (void*)Qb,  nullptr);
  gemm_wmma<0><<<gg, 256, 0, stream>>>(Xb, Wkb, (void*)Kb,  nullptr);
  gemm_wmma<1><<<gg, 256, 0, stream>>>(Xb, Wvb, (void*)VTb, nullptr);

  attn_kernel<<<Bc * Hc * (Sc / 16), 32, 0, stream>>>(Qb, Kb, VTb, CTXb);

  gemm_wmma<2><<<gg, 256, 0, stream>>>(CTXb, Wob, d_out, bo);
}
